// CapsNet_78967268704466
// MI455X (gfx1250) — matrix operand order repaired and checked by reading, not compiled, not run on hardware
//
#include <hip/hip_runtime.h>

typedef unsigned short u16t;
typedef __attribute__((ext_vector_type(16))) __bf16 v16bf;
typedef __attribute__((ext_vector_type(8)))  float  v8f;
typedef __attribute__((ext_vector_type(4)))  int    v4i;

#define BATCH   4096
#define NROUTE  512

// ---------------- bf16 helpers ----------------
__device__ __forceinline__ u16t f2bf(float f) {
  unsigned u = __builtin_bit_cast(unsigned, f);
  unsigned r = u + 0x7FFFu + ((u >> 16) & 1u);
  return (u16t)(r >> 16);
}
__device__ __forceinline__ float bf2f(u16t h) {
  unsigned u = ((unsigned)h) << 16;
  return __builtin_bit_cast(float, u);
}

// ---------------- CDNA5 async global->LDS (ASYNCcnt-tracked DMA) ----------
#if defined(__gfx1250__) && __has_builtin(__builtin_amdgcn_global_load_async_to_lds_b128)
#define HAVE_ASYNC_LDS 1
#endif

typedef __attribute__((address_space(1))) v4i GV4;   // global int4
typedef __attribute__((address_space(3))) v4i LV4;   // LDS int4

__device__ __forceinline__ void cp_b128(void* lds_dst, const void* gsrc) {
#ifdef HAVE_ASYNC_LDS
  __builtin_amdgcn_global_load_async_to_lds_b128((GV4*)(void*)gsrc,
                                                 (LV4*)lds_dst, 0, 0);
#else
  *(uint4*)lds_dst = *(const uint4*)gsrc;
#endif
}
__device__ __forceinline__ void cp_wait() {
#ifdef HAVE_ASYNC_LDS
#if __has_builtin(__builtin_amdgcn_s_wait_asynccnt)
  __builtin_amdgcn_s_wait_asynccnt(0);
#else
  asm volatile("s_wait_asynccnt 0" ::: "memory");
#endif
#endif
}

// ---------------- WMMA fragment loaders (bf16, 16x16x32) ----------------
// A (16x32, MxK): lane 0-15 -> m=lane; VGPR p holds K pair {2p,2p+1}, lane-half adds K+16
__device__ __forceinline__ v16bf load_frag_A(const u16t* p) {
  union { v16bf v; unsigned u[8]; } f;
#pragma unroll
  for (int i = 0; i < 8; ++i) {
    int k = 2 * i;
    f.u[i] = *(const unsigned*)(p + k);
  }
  return f.v;
}
// B (32x16, KxN) staged transposed in LDS as Bt[n][k]: lane = column n,
// VGPR p holds K pair {2p,2p+1}, lane-half adds K+16.
__device__ __forceinline__ v16bf load_frag_B(const u16t* p) {
  union { v16bf v; unsigned u[8]; } f;
#pragma unroll
  for (int i = 0; i < 8; ++i) f.u[i] = *(const unsigned*)(p + 2 * i);
  return f.v;
}
__device__ __forceinline__ v16bf zero_frag() {
  union { v16bf v; unsigned u[8]; } f;
#pragma unroll
  for (int i = 0; i < 8; ++i) f.u[i] = 0u;
  return f.v;
}
__device__ __forceinline__ v8f wmma_bf16(v16bf a, v16bf b, v8f c) {
  return __builtin_amdgcn_wmma_f32_16x16x32_bf16(false, a, false, b, (short)0, c, false, false);
}

// ---------------- Kernel 0: weight repack fp32 -> bf16, pre-transposed ----
// prim_w [oc=64][ic=256][3][3] -> wT2[tap=9][oc=64][ic=256]  (B^T layout)
// dec1 [64][512]->[512][64], dec2 [512][1024]->[1024][512],
// dec3 [1024][64]->[64][1024]   (all stored N-major = B^T, contiguous K rows)
__global__ __launch_bounds__(256) void repack_kernel(
    const float* __restrict__ prim_w, const float* __restrict__ d1,
    const float* __restrict__ d2, const float* __restrict__ d3,
    u16t* __restrict__ wT2, u16t* __restrict__ d1t, u16t* __restrict__ d2t,
    u16t* __restrict__ d3t) {
  int i = blockIdx.x * 256 + threadIdx.x;
  if (i < 147456) {                       // 9*64*256, [kk][oc][ic]
    int kk = i >> 14, rem = i & 16383;
    int oc = rem >> 8, ic = rem & 255;
    wT2[i] = f2bf(prim_w[((size_t)oc * 256 + ic) * 9 + kk]);
  } else if (i < 147456 + 32768) {        // [n=512][k=64]
    int j = i - 147456, n = j >> 6, k = j & 63;
    d1t[j] = f2bf(d1[k * 512 + n]);
  } else if (i < 147456 + 32768 + 524288) { // [n=1024][k=512]
    int j = i - 147456 - 32768, n = j >> 9, k = j & 511;
    d2t[j] = f2bf(d2[k * 1024 + n]);
  } else if (i < 147456 + 32768 + 524288 + 65536) { // [n=64][k=1024]
    int j = i - 147456 - 32768 - 524288, n = j >> 10, k = j & 1023;
    d3t[j] = f2bf(d3[k * 64 + n]);
  }
}

// ---------------- Kernel 1: fused conv1 + PrimaryCaps WMMA + squash -------
// One block per image. conv1 (1->256ch, 3x3) computed in-LDS (thread = ic,
// 9 weights in registers), then implicit-GEMM over 9 taps x 256 ic on WMMA,
// then per-lane squash (VGPR index g == capsule dim w).
#define XS 264   // LDS row stride in halves: 528B = 16B-aligned, 132-word banks
__global__ __launch_bounds__(256) void prim_caps_kernel(
    const float* __restrict__ data, const float* __restrict__ c1w,
    const float* __restrict__ c1b, const u16t* __restrict__ wT2,
    const float* __restrict__ prim_b, u16t* __restrict__ u_out) {
  extern __shared__ u16t smem[];
  u16t* xs  = smem;            // [64 pos][264] conv1 activations (bf16)
  u16t* bts = smem + 64 * XS;  // [64 oc][264] tap weights B^T
  __shared__ float dimg[64];
  int b = blockIdx.x;
  int tid = threadIdx.x;
  int w = tid >> 5, lane = tid & 31;
  int r = lane & 15, g = lane >> 4;
  int mt = w >> 1;             // 0..3  (M tile: 16 positions)
  int ntb = (w & 1) * 2;       // N tiles {ntb, ntb+1}

  // --- conv1: image (64 floats) to LDS, weights to registers ---
  if (tid < 64) dimg[tid] = data[(size_t)b * 64 + tid];
  float wreg[9];
#pragma unroll
  for (int t = 0; t < 9; ++t) wreg[t] = c1w[tid * 9 + t];
  float bia1 = c1b[tid];
  __syncthreads();
#pragma unroll 4
  for (int pos = 0; pos < 64; ++pos) {
    int hh = pos >> 3, ww = pos & 7;
    float acc = bia1;
#pragma unroll
    for (int ky = 0; ky < 3; ++ky)
#pragma unroll
      for (int kx = 0; kx < 3; ++kx) {
        int y = hh + ky - 1, x = ww + kx - 1;
        if (y >= 0 && y < 8 && x >= 0 && x < 8)
          acc = fmaf(dimg[y * 8 + x], wreg[ky * 3 + kx], acc);
      }
    xs[pos * XS + tid] = f2bf(fmaxf(acc, 0.f));
  }

  v8f acc0 = {}, acc1 = {};
#pragma unroll 1
  for (int kk = 0; kk < 9; ++kk) {
    __syncthreads();
    // stage tap-weight tile (already transposed [oc][ic]) via async DMA
    const u16t* wk = wT2 + kk * 16384;   // 16384 halves = 2048 b128
#pragma unroll
    for (int j = 0; j < 8; ++j) {
      int idx = (tid + j * 256) * 8;     // half offset, 16B granules
      cp_b128(bts + (idx >> 8) * XS + (idx & 255), wk + idx);
    }
    cp_wait();
    __syncthreads();

    int ky = kk / 3, kx = kk % 3;
    int pos = mt * 16 + r;               // output position (row m of A)
    int hs = (pos >> 3) + ky - 1, ws = (pos & 7) + kx - 1;
    bool valid = (hs >= 0) & (hs < 8) & (ws >= 0) & (ws < 8);
    const u16t* arow  = xs + (valid ? (hs * 8 + ws) : 0) * XS + g * 16;
    const u16t* brow0 = bts + (ntb * 16 + r) * XS + g * 16;
    const u16t* brow1 = bts + ((ntb + 1) * 16 + r) * XS + g * 16;
#pragma unroll
    for (int ks = 0; ks < 8; ++ks) {
      int kb = ks * 32;
      v16bf a  = valid ? load_frag_A(arow + kb) : zero_frag();
      v16bf b0 = load_frag_B(brow0 + kb);
      v16bf b1 = load_frag_B(brow1 + kb);
      acc0 = wmma_bf16(a, b0, acc0);
      acc1 = wmma_bf16(a, b1, acc1);
    }
  }

  // Fused bias + squash: per lane, VGPR index g == capsule dim w (8 values).
  int h = mt * 2 + g;
#pragma unroll
  for (int t = 0; t < 2; ++t) {
    v8f acc = t ? acc1 : acc0;
    int oc = (ntb + t) * 16 + r;
    float bia = prim_b[oc];
    float vals[8]; float sq = 0.f;
#pragma unroll
    for (int gg = 0; gg < 8; ++gg) { vals[gg] = acc[gg] + bia; sq += vals[gg] * vals[gg]; }
    float scale = sq / ((1.f + sq) * sqrtf(sq + 1e-9f));
    union { uint4 q; u16t s[8]; } pk;
#pragma unroll
    for (int gg = 0; gg < 8; ++gg) pk.s[gg] = f2bf(vals[gg] * scale);
    int route = oc * 8 + h;
    *(uint4*)(u_out + ((size_t)b * NROUTE + route) * 8) = pk.q;
  }
}

// ---------------- Kernel 2: zero routing logits ----------------
__global__ void zero_kernel(float* p, int n) {
  int i = blockIdx.x * blockDim.x + threadIdx.x;
  if (i < n) p[i] = 0.f;
}

// ---------------- Kernel 3: softmax over routes + copy logits -------------
__global__ __launch_bounds__(512) void softmax_kernel(
    const float* __restrict__ bin, float* __restrict__ cbuf,
    float* __restrict__ bout) {
  __shared__ float red[512];
  int t = threadIdx.x;
  float v0 = bin[t * 2 + 0], v1 = bin[t * 2 + 1];
  red[t] = v0; __syncthreads();
  for (int s = 256; s > 0; s >>= 1) { if (t < s) red[t] = fmaxf(red[t], red[t + s]); __syncthreads(); }
  float m0 = red[0]; __syncthreads();
  red[t] = v1; __syncthreads();
  for (int s = 256; s > 0; s >>= 1) { if (t < s) red[t] = fmaxf(red[t], red[t + s]); __syncthreads(); }
  float m1 = red[0]; __syncthreads();
  float e0 = __expf(v0 - m0), e1 = __expf(v1 - m1);
  red[t] = e0; __syncthreads();
  for (int s = 256; s > 0; s >>= 1) { if (t < s) red[t] += red[t + s]; __syncthreads(); }
  float s0 = red[0]; __syncthreads();
  red[t] = e1; __syncthreads();
  for (int s = 256; s > 0; s >>= 1) { if (t < s) red[t] += red[t + s]; __syncthreads(); }
  float s1 = red[0];
  cbuf[t * 2 + 0] = e0 / s0;
  cbuf[t * 2 + 1] = e1 / s1;
  bout[t * 2 + 0] = v0;
  bout[t * 2 + 1] = v1;
}

// ---------------- Kernel 4: per-image routing step ------------------------
__global__ __launch_bounds__(256) void route_kernel(
    const u16t* __restrict__ u, const float* __restrict__ W,
    const float* __restrict__ cbuf, float* __restrict__ vj,
    float* __restrict__ abuf, int doAgree) {
  __shared__ u16t us[NROUTE * 8];
  __shared__ float sp[4][64];
  __shared__ float sj[64];
  __shared__ float scale[2];
  __shared__ float vsh[64];
  int b = blockIdx.x, tid = threadIdx.x;

  { // async copy of u[b] (4096 halves = 512 b128)
    const u16t* src = u + (size_t)b * 4096;
    cp_b128(us + tid * 8, src + tid * 8);
    cp_b128(us + (tid + 256) * 8, src + (tid + 256) * 8);
    cp_wait();
  }
  __syncthreads();

  int od = tid & 63, o = od >> 5, d = od & 31, rg = tid >> 6;
  float acc = 0.f;
  for (int r = rg; r < NROUTE; r += 4) {
    const float* Wp = W + (((size_t)r * 2 + o) * 32 + d) * 8;
    float uh = 0.f;
#pragma unroll
    for (int i = 0; i < 8; ++i) uh = fmaf(Wp[i], bf2f(us[r * 8 + i]), uh);
    acc = fmaf(cbuf[r * 2 + o], uh, acc);
  }
  sp[rg][od] = acc;
  __syncthreads();
  if (tid < 64) sj[tid] = sp[0][tid] + sp[1][tid] + sp[2][tid] + sp[3][tid];
  __syncthreads();
  if (tid < 2) {
    float sq = 0.f;
    for (int dd = 0; dd < 32; ++dd) { float v = sj[tid * 32 + dd]; sq += v * v; }
    scale[tid] = sq / ((1.f + sq) * sqrtf(sq + 1e-9f));
  }
  __syncthreads();
  if (tid < 64) {
    float v = sj[tid] * scale[tid >> 5];
    vj[(size_t)b * 64 + tid] = v;
    vsh[tid] = v;
  }
  __syncthreads();
  if (doAgree) {
    for (int rr = tid; rr < NROUTE; rr += 256) {
#pragma unroll
      for (int oo = 0; oo < 2; ++oo) {
        float a = 0.f;
        const float* Wp = W + (((size_t)rr * 2 + oo) * 32) * 8;
        for (int dd = 0; dd < 32; ++dd) {
          float uh = 0.f;
#pragma unroll
          for (int i = 0; i < 8; ++i) uh = fmaf(Wp[dd * 8 + i], bf2f(us[rr * 8 + i]), uh);
          a = fmaf(uh, vsh[oo * 32 + dd], a);
        }
        abuf[(size_t)b * 1024 + rr * 2 + oo] = a;
      }
    }
  }
}

// ---------------- Kernel 5: deterministic batch-mean of agreement ---------
__global__ __launch_bounds__(256) void mean_reduce_kernel(
    const float* __restrict__ abuf, float* __restrict__ bout) {
  __shared__ float red[256];
  int col = blockIdx.x, t = threadIdx.x;
  float s = 0.f;
  for (int b = t; b < BATCH; b += 256) s += abuf[(size_t)b * 1024 + col];
  red[t] = s; __syncthreads();
  for (int k = 128; k > 0; k >>= 1) { if (t < k) red[t] += red[t + k]; __syncthreads(); }
  if (t == 0) bout[col] += red[0] * (1.f / (float)BATCH);
}

// ---------------- Kernel 6: argmax mask + outputs + decoder input ---------
__global__ __launch_bounds__(64) void mask_kernel(
    const float* __restrict__ vj, float* __restrict__ out,
    u16t* __restrict__ h) {
  __shared__ float vsh[64];
  __shared__ float cls[2];
  int b = blockIdx.x, t = threadIdx.x;
  float v = vj[(size_t)b * 64 + t];
  vsh[t] = v; __syncthreads();
  if (t < 2) {
    float s = 0.f;
    for (int dd = 0; dd < 32; ++dd) { float x = vsh[t * 32 + dd]; s += x * x; }
    cls[t] = sqrtf(s);
  }
  __syncthreads();
  int idx = cls[1] > cls[0] ? 1 : 0;
  out[(size_t)b * 64 + t] = v;                                              // output
  if (t < 2) out[(size_t)BATCH * 128 + b * 2 + t] = (t == idx) ? 1.f : 0.f; // masked
  h[(size_t)b * 64 + t] = f2bf(((t >> 5) == idx) ? v : 0.f);
}

// ---------------- Kernel 7: generic bf16 WMMA GEMM (decoder) --------------
// C[M,N] = act(A[M,K] * Bt[N,K]^T + bias). Tile 128x64, 8 waves x (2x2).
// Bt is pre-transposed (N-major) so both LDS stages are b128-contiguous.
#define GBM 128
#define GBN 64
#define GBK 32
#define GST 40   // row stride halves: 80B = 16B-aligned, 20-word bank stride
__global__ __launch_bounds__(256) void gemm_kernel(
    const u16t* __restrict__ A, const u16t* __restrict__ Bt,
    const float* __restrict__ bias, u16t* __restrict__ outbf,
    float* __restrict__ outf, int M, int N, int K, int act) {
  __shared__ u16t As[GBM * GST];
  __shared__ u16t Bs[GBN * GST];
  int bm = blockIdx.x * GBM, bn = blockIdx.y * GBN;
  int tid = threadIdx.x, w = tid >> 5, lane = tid & 31;
  int r = lane & 15, g = lane >> 4;
  int mtb = (w & 3) * 2;
  int ntb = (w >> 2) * 2;
  v8f acc[2][2] = {};

  for (int k0 = 0; k0 < K; k0 += GBK) {
    __syncthreads();
    { // A tile: 128x32 halves, 2 b128 per thread (contiguous rows)
      int row = tid >> 1, c0 = (tid & 1) * 16;
      const u16t* src = A + (size_t)(bm + row) * K + k0 + c0;
      u16t* dst = As + row * GST + c0;
      cp_b128(dst, src);
      cp_b128(dst + 8, src + 8);
    }
    { // B tile: 64x32 halves from N-major Bt, 1 b128 per thread
      int n = tid >> 2, c = (tid & 3) * 8;
      cp_b128(Bs + n * GST + c, Bt + (size_t)(bn + n) * K + k0 + c);
    }
    cp_wait();
    __syncthreads();
    v16bf af0 = load_frag_A(As + (mtb * 16 + r) * GST + g * 16);
    v16bf af1 = load_frag_A(As + ((mtb + 1) * 16 + r) * GST + g * 16);
    v16bf bf0 = load_frag_B(Bs + (ntb * 16 + r) * GST + g * 16);
    v16bf bf1 = load_frag_B(Bs + ((ntb + 1) * 16 + r) * GST + g * 16);
    acc[0][0] = wmma_bf16(af0, bf0, acc[0][0]);
    acc[0][1] = wmma_bf16(af0, bf1, acc[0][1]);
    acc[1][0] = wmma_bf16(af1, bf0, acc[1][0]);
    acc[1][1] = wmma_bf16(af1, bf1, acc[1][1]);
  }

#pragma unroll
  for (int i = 0; i < 2; ++i)
#pragma unroll
    for (int j = 0; j < 2; ++j) {
      int n = bn + (ntb + j) * 16 + r;
      float bia = bias[n];
#pragma unroll
      for (int gg = 0; gg < 8; ++gg) {
        int m = bm + (mtb + i) * 16 + gg + 8 * g;
        float val = acc[i][j][gg] + bia;
        if (act == 0) {
          outbf[(size_t)m * N + n] = f2bf(fmaxf(val, 0.f));
        } else {
          outf[(size_t)m * N + n] = 1.f / (1.f + __expf(-val));
        }
      }
    }
}

// ---------------- launcher ----------------
extern "C" void kernel_launch(void* const* d_in, const int* in_sizes, int n_in,
                              void* d_out, int out_size, void* d_ws, size_t ws_size,
                              hipStream_t stream) {
  const float* data    = (const float*)d_in[0];
  const float* conv1_w = (const float*)d_in[1];
  const float* conv1_b = (const float*)d_in[2];
  const float* prim_w  = (const float*)d_in[3];
  const float* prim_b  = (const float*)d_in[4];
  const float* W_digit = (const float*)d_in[5];
  const float* dec1_w  = (const float*)d_in[6];
  const float* dec1_b  = (const float*)d_in[7];
  const float* dec2_w  = (const float*)d_in[8];
  const float* dec2_b  = (const float*)d_in[9];
  const float* dec3_w  = (const float*)d_in[10];
  const float* dec3_b  = (const float*)d_in[11];
  float* out = (float*)d_out;

  char* ws = (char*)d_ws;
  size_t off = 0;
  auto alloc = [&](size_t bytes) -> char* {
    char* p = ws + off;
    off = (off + bytes + 255) & ~(size_t)255;
    return p;
  };
  u16t*  wT2   = (u16t*)alloc(9 * 64 * 256 * 2);            // prim B^T bf16
  u16t*  d1t   = (u16t*)alloc(512 * 64 * 2);
  u16t*  d2t   = (u16t*)alloc(1024 * 512 * 2);
  u16t*  d3t   = (u16t*)alloc(64 * 1024 * 2);
  u16t*  usq   = (u16t*)alloc((size_t)BATCH * NROUTE * 8 * 2);
  float* vj    = (float*)alloc((size_t)BATCH * 64 * 4);
  float* bA    = (float*)alloc(1024 * 4);
  float* bB    = (float*)alloc(1024 * 4);
  float* cbuf  = (float*)alloc(1024 * 4);
  float* abuf  = (float*)alloc((size_t)BATCH * 1024 * 4);
  u16t*  hbf   = (u16t*)alloc((size_t)BATCH * 64 * 2);
  u16t*  h1bf  = (u16t*)alloc((size_t)BATCH * 512 * 2);
  u16t*  h2bf  = (u16t*)alloc((size_t)BATCH * 1024 * 2);

  // 0) weight repack (bf16 + transpose)
  repack_kernel<<<3008, 256, 0, stream>>>(prim_w, dec1_w, dec2_w, dec3_w,
                                          wT2, d1t, d2t, d3t);
  // 1) fused conv1 + primary caps WMMA + squash
  prim_caps_kernel<<<BATCH, 256, 2 * 64 * XS * sizeof(u16t), stream>>>(
      data, conv1_w, conv1_b, wT2, prim_b, usq);
  // 2) routing (3 iterations, ping-pong logits bA/bB)
  zero_kernel<<<4, 256, 0, stream>>>(bA, 1024);
  softmax_kernel<<<1, 512, 0, stream>>>(bA, cbuf, bB);
  route_kernel<<<BATCH, 256, 0, stream>>>(usq, W_digit, cbuf, vj, abuf, 1);
  mean_reduce_kernel<<<1024, 256, 0, stream>>>(abuf, bB);
  softmax_kernel<<<1, 512, 0, stream>>>(bB, cbuf, bA);
  route_kernel<<<BATCH, 256, 0, stream>>>(usq, W_digit, cbuf, vj, abuf, 1);
  mean_reduce_kernel<<<1024, 256, 0, stream>>>(abuf, bA);
  softmax_kernel<<<1, 512, 0, stream>>>(bA, cbuf, bB);
  route_kernel<<<BATCH, 256, 0, stream>>>(usq, W_digit, cbuf, vj, abuf, 0);
  // 3) mask + output/masked sections + decoder input
  mask_kernel<<<BATCH, 64, 0, stream>>>(vj, out, hbf);
  // 4) decoder GEMMs on WMMA
  gemm_kernel<<<dim3(BATCH / GBM, 512 / GBN), 256, 0, stream>>>(
      hbf, d1t, dec1_b, h1bf, nullptr, BATCH, 512, 64, 0);
  gemm_kernel<<<dim3(BATCH / GBM, 1024 / GBN), 256, 0, stream>>>(
      h1bf, d2t, dec2_b, h2bf, nullptr, BATCH, 1024, 512, 0);
  gemm_kernel<<<dim3(BATCH / GBM, 64 / GBN), 256, 0, stream>>>(
      h2bf, d3t, dec3_b, nullptr, out + (size_t)BATCH * 64, BATCH, 64, 1024, 1);
}
